// DWBlock_14087492731132
// MI455X (gfx1250) — 8-bit matrix operand order repaired and checked by reading, compiled, not run on hardware
//
#include <hip/hip_runtime.h>
#include <stdint.h>

typedef __attribute__((ext_vector_type(8))) int v8i;
typedef __attribute__((ext_vector_type(4))) int v4i;
typedef __attribute__((ext_vector_type(2))) int v2i;

#define BN_EPS 1e-5f
#define HW 784
#define IMW 28

__device__ __forceinline__ int8_t sgn8(float v) {
    return (int8_t)((v > 0.f) ? 1 : ((v < 0.f) ? -1 : 0));
}

__device__ __forceinline__ void async_g2l_b128(const int8_t* gsrc, int8_t* lds_dst) {
    uint64_t gaddr = (uint64_t)(uintptr_t)gsrc;
    uint32_t laddr = (uint32_t)(uintptr_t)lds_dst;
    asm volatile("global_load_async_to_lds_b128 %0, %1, off"
                 :: "v"(laddr), "v"(gaddr)
                 : "memory");
}

__device__ __forceinline__ void wait_async0() {
    asm volatile("s_wait_asynccnt 0x0" ::: "memory");
}

__global__ void k_binarize(const float* __restrict__ x, int8_t* __restrict__ act,
                           int C, int total) {
    int i = blockIdx.x * blockDim.x + threadIdx.x;
    if (i >= total) return;
    int p = i % HW;
    int t = i / HW;
    int c = t % C;
    int n = t / C;
    act[((size_t)(n * HW + p)) * C + c] = sgn8(x[i]);
}

__global__ void k_prep_w(const float* __restrict__ w, int8_t* __restrict__ Bp,
                         float* __restrict__ scale, float* __restrict__ bias,
                         const float* __restrict__ g, const float* __restrict__ b,
                         const float* __restrict__ m, const float* __restrict__ v,
                         int O, int K) {
    int o = blockIdx.x * blockDim.x + threadIdx.x;
    if (o >= O) return;
    const float* wr = w + (size_t)o * K;
    float s = 0.f;
    for (int k = 0; k < K; ++k) s += fabsf(wr[k]);
    float alpha = s / (float)K;
    float inv = g[o] * rsqrtf(v[o] + BN_EPS);
    scale[o] = inv * alpha;
    bias[o]  = b[o] - m[o] * inv;
    for (int k = 0; k < K; ++k)
        Bp[(((size_t)(k >> 6)) * O + o) * 64 + (k & 63)] = sgn8(wr[k]);
}

__global__ void k_prep_w1(const float* __restrict__ w1, int8_t* __restrict__ Bp1,
                          float* __restrict__ scale1, float* __restrict__ bias1,
                          const float* __restrict__ g, const float* __restrict__ b,
                          const float* __restrict__ m, const float* __restrict__ v) {
    int o = blockIdx.x * blockDim.x + threadIdx.x;
    if (o >= 256) return;
    const float* wr = w1 + (size_t)o * 144;
    float s = 0.f;
    for (int k = 0; k < 144; ++k) s += fabsf(wr[k]);
    float alpha = s / 144.f;
    float inv = g[o] * rsqrtf(v[o] + BN_EPS);
    scale1[o] = inv * alpha;
    bias1[o]  = b[o] - m[o] * inv;
    int grp = o >> 4, oc = o & 15;
    for (int k = 0; k < 192; ++k) {
        int8_t sv = (k < 144) ? sgn8(wr[k]) : (int8_t)0;
        Bp1[(((size_t)(grp * 3 + (k >> 6))) * 16 + oc) * 64 + (k & 63)] = sv;
    }
}

__global__ void __launch_bounds__(256) k_conv1(
    const int8_t* __restrict__ act0, const int8_t* __restrict__ Bp1,
    const float* __restrict__ scale1, const float* __restrict__ bias1,
    const float* __restrict__ x, int8_t* __restrict__ act1) {
    __shared__ int8_t smem[128 * 192];
    __shared__ int8_t smemB[3 * 16 * 64];
    const int C = 256;
    const int mBase = blockIdx.x * 128;
    const int grp   = blockIdx.y;
    const int tid   = threadIdx.x;

    if (tid < 192) {
        const int8_t* src = Bp1 + ((size_t)grp * 3 * 16) * 64 + tid * 16;
        async_g2l_b128(src, &smemB[tid * 16]);
    }

    {
        int r = tid >> 1, hb = tid & 1;
        int mm = mBase + r;
        int n = mm / HW, p = mm % HW, h = p / IMW, wcol = p % IMW;
        int8_t* dst = smem + r * 192 + hb * 96;
        const int8_t* src = act0 + (size_t)n * HW * C + grp * 16;
        for (int kk = 0; kk < 96; ++kk) {
            int k = hb * 96 + kk;
            int8_t val = 0;
            if (k < 144) {
                int ci = k / 9, rem = k - ci * 9, kh = rem / 3, kw = rem - kh * 3;
                int hh = h + kh - 1, ww = wcol + kw - 1;
                if (hh >= 0 && hh < IMW && ww >= 0 && ww < IMW)
                    val = src[(size_t)(hh * IMW + ww) * C + ci];
            }
            dst[kk] = val;
        }
    }
    wait_async0();
    __syncthreads();

    const int wave = tid >> 5, lane = tid & 31;
    const int halfL = lane >> 4, lm = lane & 15;

    v8i acc = (v8i){0, 0, 0, 0, 0, 0, 0, 0};
#pragma unroll
    for (int kt = 0; kt < 3; ++kt) {
        const int8_t* ap = smem + (wave * 16 + lm) * 192 + kt * 64 + halfL * 16;
        v4i a0 = *(const v4i*)(ap + 0);
        v4i a1 = *(const v4i*)(ap + 32);
        v8i A;
        A[0] = a0.x; A[1] = a0.y; A[2] = a0.z; A[3] = a0.w;
        A[4] = a1.x; A[5] = a1.y; A[6] = a1.z; A[7] = a1.w;
        const int8_t* bp = smemB + (kt * 16 + lm) * 64 + halfL * 16;
        v4i b0 = *(const v4i*)(bp + 0);
        v4i b1 = *(const v4i*)(bp + 32);
        v8i B;
        B[0] = b0.x; B[1] = b0.y; B[2] = b0.z; B[3] = b0.w;
        B[4] = b1.x; B[5] = b1.y; B[6] = b1.z; B[7] = b1.w;
        acc = __builtin_amdgcn_wmma_i32_16x16x64_iu8(true, A, true, B, acc, false, false);
    }

    const int c = grp * 16 + lm;
    const float sc = scale1[c], bi = bias1[c];
#pragma unroll
    for (int vg = 0; vg < 8; ++vg) {
        int mm = mBase + wave * 16 + vg + 8 * halfL;
        int n = mm / HW, p = mm % HW;
        float y = (float)acc[vg] * sc + bi + x[((size_t)n * C + c) * HW + p];
        y = fminf(1.f, fmaxf(-1.f, y));
        act1[(size_t)mm * C + c] = sgn8(y);
    }
}

__global__ void __launch_bounds__(256) k_conv2(
    const int8_t* __restrict__ actA, const int8_t* __restrict__ Bp,
    const float* __restrict__ scale, const float* __restrict__ bias,
    float* __restrict__ res, int8_t* __restrict__ act2) {
    const int K = 256, N = 512;
    __shared__ int8_t smemB[4 * 64 * 64];
    const int mBase = blockIdx.x * 128;
    const int nBase = blockIdx.y * 64;
    const int tid = threadIdx.x, wave = tid >> 5, lane = tid & 31;
    const int halfL = lane >> 4, lm = lane & 15;

#pragma unroll
    for (int j = 0; j < 4; ++j) {
        int i = tid + 256 * j;
        int kt = i >> 8, rem = i & 255;
        const int8_t* src = Bp + ((size_t)kt * N + nBase) * 64 + rem * 16;
        async_g2l_b128(src, &smemB[kt * 4096 + rem * 16]);
    }

    const int mRow = mBase + wave * 16 + lm;
    const int8_t* arow = actA + (size_t)mRow * K;

    v8i acc[4];
#pragma unroll
    for (int i = 0; i < 4; ++i) acc[i] = (v8i){0, 0, 0, 0, 0, 0, 0, 0};

    wait_async0();
    __syncthreads();

#pragma unroll
    for (int kt = 0; kt < K / 64; ++kt) {
        if (kt + 1 < K / 64) __builtin_prefetch(arow + (kt + 1) * 64, 0, 0);
        const int8_t* ap = arow + kt * 64 + halfL * 16;
        v4i a0 = *(const v4i*)(ap + 0);
        v4i a1 = *(const v4i*)(ap + 32);
        v8i A;
        A[0] = a0.x; A[1] = a0.y; A[2] = a0.z; A[3] = a0.w;
        A[4] = a1.x; A[5] = a1.y; A[6] = a1.z; A[7] = a1.w;
#pragma unroll
        for (int nt = 0; nt < 4; ++nt) {
            const int8_t* bp = smemB + kt * 4096 + (nt * 16 + lm) * 64 + halfL * 16;
            v4i b0 = *(const v4i*)(bp + 0);
            v4i b1 = *(const v4i*)(bp + 32);
            v8i B;
            B[0] = b0.x; B[1] = b0.y; B[2] = b0.z; B[3] = b0.w;
            B[4] = b1.x; B[5] = b1.y; B[6] = b1.z; B[7] = b1.w;
            acc[nt] = __builtin_amdgcn_wmma_i32_16x16x64_iu8(true, A, true, B, acc[nt], false, false);
        }
    }

#pragma unroll
    for (int nt = 0; nt < 4; ++nt) {
        int col = nBase + nt * 16 + lm;
        float sc = scale[col], bi = bias[col];
#pragma unroll
        for (int vg = 0; vg < 8; ++vg) {
            int mm = mBase + wave * 16 + vg + 8 * halfL;
            float y = (float)acc[nt][vg] * sc + bi;
            y = fminf(1.f, fmaxf(-1.f, y));
            res[(size_t)mm * N + col] = y;
            act2[(size_t)mm * N + col] = sgn8(y);
        }
    }
}

__global__ void __launch_bounds__(256) k_conv3(
    const int8_t* __restrict__ actA, const int8_t* __restrict__ Bp,
    const float* __restrict__ scale, const float* __restrict__ bias,
    const float* __restrict__ res, float* __restrict__ out) {
    const int K = 512, N = 512;
    __shared__ int8_t smemB[8 * 64 * 64];
    const int mBase = blockIdx.x * 128;
    const int nBase = blockIdx.y * 64;
    const int tid = threadIdx.x, wave = tid >> 5, lane = tid & 31;
    const int halfL = lane >> 4, lm = lane & 15;

#pragma unroll
    for (int j = 0; j < 8; ++j) {
        int i = tid + 256 * j;
        int kt = i >> 8, rem = i & 255;
        const int8_t* src = Bp + ((size_t)kt * N + nBase) * 64 + rem * 16;
        async_g2l_b128(src, &smemB[kt * 4096 + rem * 16]);
    }

    const int mRow = mBase + wave * 16 + lm;
    const int8_t* arow = actA + (size_t)mRow * K;

    v8i acc[4];
#pragma unroll
    for (int i = 0; i < 4; ++i) acc[i] = (v8i){0, 0, 0, 0, 0, 0, 0, 0};

    wait_async0();
    __syncthreads();

#pragma unroll
    for (int kt = 0; kt < K / 64; ++kt) {
        if (kt + 1 < K / 64) __builtin_prefetch(arow + (kt + 1) * 64, 0, 0);
        const int8_t* ap = arow + kt * 64 + halfL * 16;
        v4i a0 = *(const v4i*)(ap + 0);
        v4i a1 = *(const v4i*)(ap + 32);
        v8i A;
        A[0] = a0.x; A[1] = a0.y; A[2] = a0.z; A[3] = a0.w;
        A[4] = a1.x; A[5] = a1.y; A[6] = a1.z; A[7] = a1.w;
#pragma unroll
        for (int nt = 0; nt < 4; ++nt) {
            const int8_t* bp = smemB + kt * 4096 + (nt * 16 + lm) * 64 + halfL * 16;
            v4i b0 = *(const v4i*)(bp + 0);
            v4i b1 = *(const v4i*)(bp + 32);
            v8i B;
            B[0] = b0.x; B[1] = b0.y; B[2] = b0.z; B[3] = b0.w;
            B[4] = b1.x; B[5] = b1.y; B[6] = b1.z; B[7] = b1.w;
            acc[nt] = __builtin_amdgcn_wmma_i32_16x16x64_iu8(true, A, true, B, acc[nt], false, false);
        }
    }

#pragma unroll
    for (int nt = 0; nt < 4; ++nt) {
        int col = nBase + nt * 16 + lm;
        float sc = scale[col], bi = bias[col];
#pragma unroll
        for (int vg = 0; vg < 8; ++vg) {
            int mm = mBase + wave * 16 + vg + 8 * halfL;
            int n = mm / HW, p = mm % HW;
            float y = (float)acc[nt][vg] * sc + bi + res[(size_t)mm * N + col];
            y = fminf(1.f, fmaxf(-1.f, y));
            out[((size_t)n * N + col) * HW + p] = y;
        }
    }
}

extern "C" void kernel_launch(void* const* d_in, const int* in_sizes, int n_in,
                              void* d_out, int out_size, void* d_ws, size_t ws_size,
                              hipStream_t stream) {
    const float* x  = (const float*)d_in[0];
    const float* w1 = (const float*)d_in[1];
    const float* w2 = (const float*)d_in[2];
    const float* w3 = (const float*)d_in[3];
    const float* g1 = (const float*)d_in[4];
    const float* b1 = (const float*)d_in[5];
    const float* m1 = (const float*)d_in[6];
    const float* v1 = (const float*)d_in[7];
    const float* g2 = (const float*)d_in[8];
    const float* b2 = (const float*)d_in[9];
    const float* m2 = (const float*)d_in[10];
    const float* v2 = (const float*)d_in[11];
    const float* g3 = (const float*)d_in[12];
    const float* b3 = (const float*)d_in[13];
    const float* m3 = (const float*)d_in[14];
    const float* v3 = (const float*)d_in[15];
    float* out = (float*)d_out;

    const int Nb = 64, Cin = 256, Cout = 512;
    const size_t M = (size_t)Nb * HW;

    uint8_t* ws = (uint8_t*)d_ws;
    size_t off = 0;
    auto wsalloc = [&](size_t bytes) -> void* {
        void* p = (void*)(ws + off);
        off = (off + bytes + 255) & ~(size_t)255;
        return p;
    };
    int8_t* act0   = (int8_t*)wsalloc(M * Cin);
    int8_t* act1   = (int8_t*)wsalloc(M * Cin);
    int8_t* act2   = (int8_t*)wsalloc(M * Cout);
    float*  resf   = (float*)wsalloc(M * Cout * 4);
    int8_t* Bp1    = (int8_t*)wsalloc((size_t)16 * 3 * 16 * 64);
    int8_t* Bp2    = (int8_t*)wsalloc((size_t)4 * Cout * 64);
    int8_t* Bp3    = (int8_t*)wsalloc((size_t)8 * Cout * 64);
    float*  scale1 = (float*)wsalloc(Cin * 4);
    float*  bias1  = (float*)wsalloc(Cin * 4);
    float*  scale2 = (float*)wsalloc(Cout * 4);
    float*  bias2  = (float*)wsalloc(Cout * 4);
    float*  scale3 = (float*)wsalloc(Cout * 4);
    float*  bias3  = (float*)wsalloc(Cout * 4);
    (void)ws_size; (void)in_sizes; (void)n_in; (void)out_size;

    {
        int total = (int)(M * Cin);
        k_binarize<<<(total + 255) / 256, 256, 0, stream>>>(x, act0, Cin, total);
    }
    k_prep_w1<<<1, 256, 0, stream>>>(w1, Bp1, scale1, bias1, g1, b1, m1, v1);
    k_prep_w<<<2, 256, 0, stream>>>(w2, Bp2, scale2, bias2, g2, b2, m2, v2, Cout, Cin);
    k_prep_w<<<2, 256, 0, stream>>>(w3, Bp3, scale3, bias3, g3, b3, m3, v3, Cout, Cout);

    k_conv1<<<dim3(392, 16), 256, 0, stream>>>(act0, Bp1, scale1, bias1, x, act1);
    k_conv2<<<dim3(392, 8), 256, 0, stream>>>(act1, Bp2, scale2, bias2, resf, act2);
    k_conv3<<<dim3(392, 8), 256, 0, stream>>>(act2, Bp3, scale3, bias3, resf, out);
}
